// HAN_90168543412642
// MI455X (gfx1250) — compile-verified, matrix operand order checked by reading, not run on hardware
//
#include <hip/hip_runtime.h>

#define N_AUTH 50000
#define N_PAP  50000
#define NEDGE  600000
#define HH 8
#define DD 16
#define CC 128
#define OUT_C 16
#define F_A 256
#define F_P 128
#define NEG_SLOPE 0.2f
#define EPS_GN 1e-5f

typedef __attribute__((ext_vector_type(16))) __bf16 v16bf;
typedef __attribute__((ext_vector_type(8)))  float  v8f;
typedef __attribute__((ext_vector_type(4)))  float  v4f;
typedef __attribute__((ext_vector_type(4)))  unsigned v4u;

// ---------- helpers ----------
__device__ __forceinline__ unsigned f32_to_bf16_rne(float f) {
  unsigned u = __float_as_uint(f);
  unsigned r = u + 0x7FFFu + ((u >> 16) & 1u);
  return r >> 16;
}
__device__ __forceinline__ unsigned pack_bf16x2(float lo, float hi) {
  return f32_to_bf16_rne(lo) | (f32_to_bf16_rne(hi) << 16);
}
// order-preserving float<->uint map (for atomicMax on floats incl. negatives)
__device__ __forceinline__ unsigned f2ord(float f) {
  unsigned u = __float_as_uint(f);
  return (u & 0x80000000u) ? ~u : (u | 0x80000000u);
}
__device__ __forceinline__ float ord2f(unsigned u) {
  return (u & 0x80000000u) ? __uint_as_float(u & 0x7FFFFFFFu) : __uint_as_float(~u);
}
__device__ __forceinline__ float leaky(float a) {
  return (a >= 0.f) ? a : NEG_SLOPE * a;
}

// ---------- zero fill ----------
__global__ void zero_kernel(float* __restrict__ p, long long n) {
  long long i = (long long)blockIdx.x * blockDim.x + threadIdx.x;
  if (i < n) p[i] = 0.f;
}

// ---------- WMMA GEMM: C[M,128] = A[M,K] @ W[K,128] (+bias), bf16 compute ----------
// LDS layout is fragment-friendly: both A and B tiles are stored "lane-major"
// (As[row][k], Bs[col][k], row stride 40 ushorts = 80B, 16B-aligned) so every
// WMMA fragment is two contiguous 16-byte chunks -> ds_load_b128.
// MODE 0: store C + bias.
// MODE 1: accumulate column-sums of tanh(C + bias) into Cout[0..127].
template <int MODE>
__global__ __launch_bounds__(256) void gemm_wmma_kernel(
    const float* __restrict__ A, const float* __restrict__ W,
    const float* __restrict__ bias, float* __restrict__ Cout, int M, int K) {
  __shared__ unsigned short As[64][40];   // 5.0 KB
  __shared__ unsigned short Bs[CC][40];   // 10.0 KB (B stored TRANSPOSED: [col][k])
  __shared__ float colacc[CC];

  unsigned* As32 = (unsigned*)&As[0][0];  // row stride 20 dwords
  unsigned* Bs32 = (unsigned*)&Bs[0][0];

  const int tid  = threadIdx.x;
  const int m0   = blockIdx.x * 64;
  const int wave = tid >> 5;
  const int lane = tid & 31;
  const int wm   = wave & 3;         // M sub-strip (16 rows)
  const int wn   = wave >> 2;        // N half (64 cols)
  const int n0w  = wn * 64;
  const int kb   = (lane >> 4) << 3; // 0 or 8: K base within fragment

  v8f acc[4];
#pragma unroll
  for (int s = 0; s < 4; ++s)
#pragma unroll
    for (int r = 0; r < 8; ++r) acc[s][r] = 0.f;

  // A staging map: thread -> (row 0..63, 8 consecutive k)
  const int a_row = tid >> 2;
  const int a_kc  = (tid & 3) * 8;
  const bool a_ok = (m0 + a_row) < M;
  // B staging map: thread -> (k-pair 0..15, 8 consecutive cols); writes transposed
  const int kpair = tid >> 4;
  const int cbase = (tid & 15) * 8;

  for (int k0 = 0; k0 < K; k0 += 32) {
    // ---- stage A tile: one branch per thread, two b128 loads, packed dword writes
    {
      v4f va = {0.f, 0.f, 0.f, 0.f}, vb = {0.f, 0.f, 0.f, 0.f};
      if (a_ok) {
        const v4f* ap = (const v4f*)(A + (size_t)(m0 + a_row) * K + k0 + a_kc);
        va = __builtin_nontemporal_load(ap);      // A is streamed once; keep L2
        vb = __builtin_nontemporal_load(ap + 1);  // for the edge-phase tables
      }
      unsigned* dst = As32 + a_row * 20 + (a_kc >> 1);
      dst[0] = pack_bf16x2(va.x, va.y);
      dst[1] = pack_bf16x2(va.z, va.w);
      dst[2] = pack_bf16x2(vb.x, vb.y);
      dst[3] = pack_bf16x2(vb.z, vb.w);
    }
    // ---- stage B tile transposed: Bs[col][k] (K multiple of 32, no guard) ----
    {
      const v4f* w0 = (const v4f*)(W + (size_t)(k0 + 2 * kpair) * CC + cbase);
      const v4f* w1 = (const v4f*)(W + (size_t)(k0 + 2 * kpair + 1) * CC + cbase);
      v4f r0a = w0[0], r0b = w0[1];
      v4f r1a = w1[0], r1b = w1[1];
      unsigned* bdst = Bs32 + kpair;
      bdst[(cbase + 0) * 20] = pack_bf16x2(r0a.x, r1a.x);
      bdst[(cbase + 1) * 20] = pack_bf16x2(r0a.y, r1a.y);
      bdst[(cbase + 2) * 20] = pack_bf16x2(r0a.z, r1a.z);
      bdst[(cbase + 3) * 20] = pack_bf16x2(r0a.w, r1a.w);
      bdst[(cbase + 4) * 20] = pack_bf16x2(r0b.x, r1b.x);
      bdst[(cbase + 5) * 20] = pack_bf16x2(r0b.y, r1b.y);
      bdst[(cbase + 6) * 20] = pack_bf16x2(r0b.z, r1b.z);
      bdst[(cbase + 7) * 20] = pack_bf16x2(r0b.w, r1b.w);
    }
    __syncthreads();

    // ---- fragments: two contiguous 16B chunks each (k = kb..kb+7, kb+16..kb+23)
    union fragu { v16bf v; v4u q[2]; } af, bf;
    const unsigned short* abase = &As[wm * 16 + (lane & 15)][kb];
    af.q[0] = *(const v4u*)(abase);
    af.q[1] = *(const v4u*)(abase + 16);
#pragma unroll
    for (int s = 0; s < 4; ++s) {
      const unsigned short* bbase = &Bs[n0w + s * 16 + (lane & 15)][kb];
      bf.q[0] = *(const v4u*)(bbase);
      bf.q[1] = *(const v4u*)(bbase + 16);
      acc[s] = __builtin_amdgcn_wmma_f32_16x16x32_bf16(
          false, af.v, false, bf.v, (short)0, acc[s], false, false);
    }
    __syncthreads();
  }

  if (MODE == 1) {
    if (tid < CC) colacc[tid] = 0.f;
    __syncthreads();
  }
  // C/D layout: VGPR r -> M = r (+8 for lanes 16-31), N = lane%16
  const int rbase = m0 + wm * 16 + ((lane >> 4) << 3);
#pragma unroll
  for (int s = 0; s < 4; ++s) {
    const int colg = n0w + s * 16 + (lane & 15);
    const float b = bias[colg];
#pragma unroll
    for (int r = 0; r < 8; ++r) {
      int row = rbase + r;
      if (row < M) {
        float v = acc[s][r] + b;
        if (MODE == 0) Cout[(size_t)row * CC + colg] = v;
        else           atomicAdd(&colacc[colg], tanhf(v));   // ds_add_f32
      }
    }
  }
  if (MODE == 1) {
    __syncthreads();
    if (tid < CC) atomicAdd(&Cout[tid], colacc[tid]);
  }
}

// ---------- per-node attention logits: alpha[n,h] = sum_d h[n,h,d]*att[h,d] ----------
__global__ void alpha_kernel(const float* __restrict__ h, const float* __restrict__ att,
                             float* __restrict__ alpha, int N) {
  int idx = blockIdx.x * blockDim.x + threadIdx.x;
  if (idx >= N * HH) return;
  int n = idx >> 3, hh = idx & 7;
  const float* hp = h + (size_t)n * CC + hh * DD;
  const float* ap = att + hh * DD;
  float s = 0.f;
#pragma unroll
  for (int d = 0; d < DD; ++d) s += hp[d] * ap[d];
  alpha[idx] = s;
}

// ---------- edge pass 1: segment max over dst (ordered-uint atomicMax) ----------
__global__ void edge_max_kernel(const int* __restrict__ edge, const float* __restrict__ as,
                                const float* __restrict__ ad, unsigned* __restrict__ mbuf) {
  int idx = blockIdx.x * blockDim.x + threadIdx.x;
  if (idx >= NEDGE * HH) return;
  int e = idx >> 3, hh = idx & 7;
  int src = edge[e], dst = edge[NEDGE + e];
  float a = leaky(as[src * HH + hh] + ad[dst * HH + hh]);
  atomicMax(&mbuf[dst * HH + hh], f2ord(a));
}

// ---------- edge pass 2: segment sum of exp(a - m) ----------
__global__ void edge_sum_kernel(const int* __restrict__ edge, const float* __restrict__ as,
                                const float* __restrict__ ad, const unsigned* __restrict__ mbuf,
                                float* __restrict__ den) {
  int idx = blockIdx.x * blockDim.x + threadIdx.x;
  if (idx >= NEDGE * HH) return;
  int e = idx >> 3, hh = idx & 7;
  int src = edge[e], dst = edge[NEDGE + e];
  float a = leaky(as[src * HH + hh] + ad[dst * HH + hh]);
  float m = ord2f(mbuf[dst * HH + hh]);
  atomicAdd(&den[dst * HH + hh], expf(a - m));
}

// ---------- edge pass 3: weighted message scatter (16 channels per thread) ----------
__global__ void edge_msg_kernel(const int* __restrict__ edge, const float* __restrict__ as,
                                const float* __restrict__ ad, const unsigned* __restrict__ mbuf,
                                const float* __restrict__ den, const float* __restrict__ hsrc,
                                float* __restrict__ outp) {
  int idx = blockIdx.x * blockDim.x + threadIdx.x;
  if (idx >= NEDGE * HH) return;
  int e = idx >> 3, hh = idx & 7;
  int src = edge[e], dst = edge[NEDGE + e];
  float a = leaky(as[src * HH + hh] + ad[dst * HH + hh]);
  float m = ord2f(mbuf[dst * HH + hh]);
  float attn = expf(a - m) / den[dst * HH + hh];
  const float* hp = hsrc + (size_t)src * CC + hh * DD;
  float* op = outp + (size_t)dst * CC + hh * DD;
#pragma unroll
  for (int d = 0; d < DD; ++d) atomicAdd(&op[d], hp[d] * attn);
}

// ---------- elementwise ReLU ----------
__global__ void relu_kernel(float* __restrict__ p, long long n) {
  long long i = (long long)blockIdx.x * blockDim.x + threadIdx.x;
  if (i < n) p[i] = fmaxf(p[i], 0.f);
}

// ---------- semantic softmax beta (1 block) ----------
__global__ __launch_bounds__(128) void beta_kernel(const float* __restrict__ sem_acc,
                                                   const float* __restrict__ q,
                                                   float* __restrict__ beta) {
  __shared__ float r0[128], r1[128];
  int t = threadIdx.x;
  float qc = q[t];
  r0[t] = sem_acc[t] * qc;
  r1[t] = sem_acc[CC + t] * qc;
  __syncthreads();
  for (int s = 64; s > 0; s >>= 1) {
    if (t < s) { r0[t] += r0[t + s]; r1[t] += r1[t + s]; }
    __syncthreads();
  }
  if (t == 0) {
    float w0 = r0[0] / (float)N_PAP, w1 = r1[0] / (float)N_PAP;
    float mx = fmaxf(w0, w1);
    float e0 = expf(w0 - mx), e1 = expf(w1 - mx);
    float inv = 1.f / (e0 + e1);
    beta[0] = e0 * inv;
    beta[1] = e1 * inv;
  }
}

// ---------- combined moments for GraphNorm (per-block partial, few atomics) ----------
__global__ __launch_bounds__(128) void combine_stats_kernel(
    const float* __restrict__ oap, const float* __restrict__ opp,
    const float* __restrict__ beta, float* __restrict__ rsum, float* __restrict__ rsq) {
  int c = threadIdx.x;
  int n0 = blockIdx.x * 32;
  float b0 = beta[0], b1 = beta[1];
  float s = 0.f, s2 = 0.f;
  for (int i = 0; i < 32; ++i) {
    int n = n0 + i;
    if (n < N_PAP) {
      size_t off = (size_t)n * CC + c;
      float v = b0 * oap[off] + b1 * opp[off];
      s += v;
      s2 += v * v;
    }
  }
  atomicAdd(&rsum[c], s);
  atomicAdd(&rsq[c], s2);
}

// ---------- GraphNorm constants: shift = mean*ms, scale = w * rsqrt(var+eps) ----------
__global__ void stats_kernel(const float* __restrict__ rsum, const float* __restrict__ rsq,
                             const float* __restrict__ ms, const float* __restrict__ nw,
                             float* __restrict__ shift, float* __restrict__ scale) {
  int c = threadIdx.x;
  if (c >= CC) return;
  float mean = rsum[c] / (float)N_PAP;
  float ex2  = rsq[c] / (float)N_PAP;
  float mu   = mean * ms[c];
  float var  = ex2 - 2.f * mu * mean + mu * mu;   // E[(v - mu)^2]
  shift[c] = mu;
  scale[c] = nw[c] * rsqrtf(var + EPS_GN);
}

// ---------- fused combine + normalize + classifier (128 -> 16) ----------
__global__ __launch_bounds__(128) void final_kernel(
    const float* __restrict__ oap, const float* __restrict__ opp,
    const float* __restrict__ beta, const float* __restrict__ shift,
    const float* __restrict__ scale, const float* __restrict__ nb,
    const float* __restrict__ linW, const float* __restrict__ linb,
    float* __restrict__ out) {
  __shared__ float Ws[CC][OUT_C];
  __shared__ float rowbuf[CC];
  __shared__ float red[128];
  int t = threadIdx.x;
#pragma unroll
  for (int o = 0; o < OUT_C; ++o) Ws[t][o] = linW[t * OUT_C + o];
  float b0 = beta[0], b1 = beta[1];
  float sh = shift[t], sc = scale[t], bb = nb[t];
  int o = t & 15, chunk = t >> 4;   // 16 outs x 8 chunks of 16 channels
  int n0 = blockIdx.x * 16;
  __syncthreads();
  for (int i = 0; i < 16; ++i) {
    int n = n0 + i;
    if (n >= N_PAP) break;          // uniform across block
    size_t off = (size_t)n * CC + t;
    float v = b0 * oap[off] + b1 * opp[off];
    rowbuf[t] = (v - sh) * sc + bb;
    __syncthreads();
    float p = 0.f;
#pragma unroll
    for (int j = 0; j < 16; ++j) {
      int c = chunk * 16 + j;
      p += rowbuf[c] * Ws[c][o];
    }
    red[t] = p;
    __syncthreads();
    if (t < 64) red[t] += red[t + 64];
    __syncthreads();
    if (t < 32) red[t] += red[t + 32];
    __syncthreads();
    if (t < 16) out[(size_t)n * OUT_C + t] = red[t] + red[t + 16] + linb[t];
    __syncthreads();
  }
}

// ---------- launch ----------
extern "C" void kernel_launch(void* const* d_in, const int* in_sizes, int n_in,
                              void* d_out, int out_size, void* d_ws, size_t ws_size,
                              hipStream_t stream) {
  const float* x_a      = (const float*)d_in[0];
  const float* x_p      = (const float*)d_in[1];
  const float* W_a      = (const float*)d_in[2];
  const float* b_a      = (const float*)d_in[3];
  const float* W_p      = (const float*)d_in[4];
  const float* b_p      = (const float*)d_in[5];
  const float* att_s_ap = (const float*)d_in[6];
  const float* att_d_ap = (const float*)d_in[7];
  const float* att_s_pp = (const float*)d_in[8];
  const float* att_d_pp = (const float*)d_in[9];
  const float* k_W      = (const float*)d_in[10];
  const float* k_b      = (const float*)d_in[11];
  const float* q        = (const float*)d_in[12];
  const float* nw       = (const float*)d_in[13];
  const float* nb       = (const float*)d_in[14];
  const float* ms       = (const float*)d_in[15];
  const float* linW     = (const float*)d_in[16];
  const float* linb     = (const float*)d_in[17];
  const int*   e_ap     = (const int*)d_in[18];
  const int*   e_pp     = (const int*)d_in[19];
  float* out = (float*)d_out;

  float* ws = (float*)d_ws;
  size_t o = 0;
  float* h_a = ws + o;   o += (size_t)N_AUTH * CC;
  float* h_p = ws + o;   o += (size_t)N_PAP * CC;
  float* as_ap = ws + o; o += (size_t)N_AUTH * HH;
  float* ad_ap = ws + o; o += (size_t)N_PAP * HH;
  float* as_pp = ws + o; o += (size_t)N_PAP * HH;
  float* ad_pp = ws + o; o += (size_t)N_PAP * HH;
  size_t zero_begin = o;
  unsigned* m_ap = (unsigned*)(ws + o); o += (size_t)N_PAP * HH;
  float* den_ap = ws + o;  o += (size_t)N_PAP * HH;
  unsigned* m_pp = (unsigned*)(ws + o); o += (size_t)N_PAP * HH;
  float* den_pp = ws + o;  o += (size_t)N_PAP * HH;
  float* out_ap = ws + o;  o += (size_t)N_PAP * CC;
  float* out_pp = ws + o;  o += (size_t)N_PAP * CC;
  float* sem_acc = ws + o; o += 2 * CC;
  float* rsum = ws + o;    o += CC;
  float* rsq = ws + o;     o += CC;
  size_t zero_end = o;
  float* beta = ws + o;    o += 2;
  float* shift = ws + o;   o += CC;
  float* scale = ws + o;   o += CC;

  // 1) zero all accumulator regions (m buffers: ord(min) == 0)
  long long nz = (long long)(zero_end - zero_begin);
  zero_kernel<<<(unsigned)((nz + 255) / 256), 256, 0, stream>>>(ws + zero_begin, nz);

  // 2) projections (WMMA bf16)
  unsigned gemm_blocks = (N_AUTH + 63) / 64;
  gemm_wmma_kernel<0><<<gemm_blocks, 256, 0, stream>>>(x_a, W_a, b_a, h_a, N_AUTH, F_A);
  gemm_wmma_kernel<0><<<gemm_blocks, 256, 0, stream>>>(x_p, W_p, b_p, h_p, N_PAP, F_P);

  // 3) per-node attention logits
  unsigned ablk = (N_AUTH * HH + 255) / 256;
  alpha_kernel<<<ablk, 256, 0, stream>>>(h_a, att_s_ap, as_ap, N_AUTH);
  alpha_kernel<<<ablk, 256, 0, stream>>>(h_p, att_d_ap, ad_ap, N_PAP);
  alpha_kernel<<<ablk, 256, 0, stream>>>(h_p, att_s_pp, as_pp, N_PAP);
  alpha_kernel<<<ablk, 256, 0, stream>>>(h_p, att_d_pp, ad_pp, N_PAP);

  // 4) edge softmax + aggregation (3 passes per metapath)
  unsigned eblk = (NEDGE * HH + 255) / 256;
  edge_max_kernel<<<eblk, 256, 0, stream>>>(e_ap, as_ap, ad_ap, m_ap);
  edge_max_kernel<<<eblk, 256, 0, stream>>>(e_pp, as_pp, ad_pp, m_pp);
  edge_sum_kernel<<<eblk, 256, 0, stream>>>(e_ap, as_ap, ad_ap, m_ap, den_ap);
  edge_sum_kernel<<<eblk, 256, 0, stream>>>(e_pp, as_pp, ad_pp, m_pp, den_pp);
  edge_msg_kernel<<<eblk, 256, 0, stream>>>(e_ap, as_ap, ad_ap, m_ap, den_ap, h_a, out_ap);
  edge_msg_kernel<<<eblk, 256, 0, stream>>>(e_pp, as_pp, ad_pp, m_pp, den_pp, h_p, out_pp);

  // 5) ReLU
  long long nrc = (long long)N_PAP * CC;
  relu_kernel<<<(unsigned)((nrc + 255) / 256), 256, 0, stream>>>(out_ap, nrc);
  relu_kernel<<<(unsigned)((nrc + 255) / 256), 256, 0, stream>>>(out_pp, nrc);

  // 6) semantic attention: fused tanh(out_r @ k_W + k_b) column-sum (WMMA bf16)
  gemm_wmma_kernel<1><<<gemm_blocks, 256, 0, stream>>>(out_ap, k_W, k_b, sem_acc, N_PAP, CC);
  gemm_wmma_kernel<1><<<gemm_blocks, 256, 0, stream>>>(out_pp, k_W, k_b, sem_acc + CC, N_PAP, CC);
  beta_kernel<<<1, 128, 0, stream>>>(sem_acc, q, beta);

  // 7) GraphNorm moments + constants
  combine_stats_kernel<<<(N_PAP + 31) / 32, 128, 0, stream>>>(out_ap, out_pp, beta, rsum, rsq);
  stats_kernel<<<1, 128, 0, stream>>>(rsum, rsq, ms, nw, shift, scale);

  // 8) fused combine + normalize + classifier
  final_kernel<<<(N_PAP + 15) / 16, 128, 0, stream>>>(out_ap, out_pp, beta, shift, scale,
                                                      nb, linW, linb, out);
}
